// BahdanauAttention_94489281076
// MI455X (gfx1250) — compile-verified, matrix operand order checked by reading, not run on hardware
//
#include <hip/hip_runtime.h>
#include <hip/hip_bf16.h>
#include <math.h>

// ---------------------------------------------------------------------------
// Bahdanau attention, MI455X (gfx1250, wave32, WMMA).
//   B=16, TQ=64, TV=256, D=512, U=512
//   out = [context (B*TQ*D f32) | attn (B*TQ*TV f32)]
// ---------------------------------------------------------------------------

#define B_  16
#define TQ_ 64
#define TV_ 256
#define D_  512
#define U_  512

typedef __bf16 bf16_t;
typedef __attribute__((ext_vector_type(16))) __bf16 v16bf;
typedef __attribute__((ext_vector_type(8)))  __bf16 v8bf;
typedef __attribute__((ext_vector_type(8)))  float  v8f;

__device__ __forceinline__ v16bf load_a_frag(const float* __restrict__ row,
                                             int k0, int half) {
    // lane holds K = {half*8 .. half*8+7} and {16+half*8 .. 16+half*8+7}:
    // two contiguous 8-float runs -> four aligned float4 loads.
    const float4 x0 = *(const float4*)(row + k0 + half * 8);
    const float4 x1 = *(const float4*)(row + k0 + half * 8 + 4);
    const float4 x2 = *(const float4*)(row + k0 + 16 + half * 8);
    const float4 x3 = *(const float4*)(row + k0 + 16 + half * 8 + 4);
    v16bf a;
    a[0]  = (bf16_t)x0.x; a[1]  = (bf16_t)x0.y; a[2]  = (bf16_t)x0.z; a[3]  = (bf16_t)x0.w;
    a[4]  = (bf16_t)x1.x; a[5]  = (bf16_t)x1.y; a[6]  = (bf16_t)x1.z; a[7]  = (bf16_t)x1.w;
    a[8]  = (bf16_t)x2.x; a[9]  = (bf16_t)x2.y; a[10] = (bf16_t)x2.z; a[11] = (bf16_t)x2.w;
    a[12] = (bf16_t)x3.x; a[13] = (bf16_t)x3.y; a[14] = (bf16_t)x3.z; a[15] = (bf16_t)x3.w;
    return a;
}

// Stage one 32(K) x 64(N) f32 slab as bf16 WMMA-B fragments into `buf`.
// Per thread: fixed column nst = t&63, k walks kb, kb+4, ... (8 iters), so the
// global pointer advances by a uniform 4*N stride — no per-element address mul.
__device__ __forceinline__ void stage_b_slab(uint32_t* __restrict__ buf,
                                             const float* __restrict__ gp0,
                                             int N, int kb, int nst) {
    const float* gp = gp0;
    const int jbase = (nst >> 4) * 2;        // 16-col tile * 2 chunks
    const int nlow  = nst & 15;
#pragma unroll
    for (int it = 0; it < 8; ++it) {
        const float f0 = gp[0];
        const float f1 = gp[(size_t)N];
        gp += 4 * (size_t)N;
        const int k  = kb + it * 4;
        const int c  = k >> 4;                        // fragment chunk
        const int wi = k & 15;
        const int l  = nlow + ((wi >> 3) << 4);       // holder lane
        union { __bf16 h[2]; uint32_t u; } pk;
        pk.h[0] = (bf16_t)f0;
        pk.h[1] = (bf16_t)f1;
        buf[(((jbase + c) * 32 + l) << 2) + ((wi & 7) >> 1)] = pk.u;
    }
}

// ---------------------------------------------------------------------------
// Generic batched GEMM + bias:  C[b] (MxN) = A[b] (MxK) @ Bg[b] (KxN) + bias
// Block = 4 waves sharing one 64-wide N column group; waves own distinct
// 16-row M tiles. Double-buffered LDS B-fragments: one barrier per K-step;
// per step: ds_load_b128 fragments + 4 WMMAs issue first, then next-slab
// global loads / cvt_pk / ds_stores overlap the matrix pipe.
// ---------------------------------------------------------------------------
#define WPB 4
#define TPB 128

__global__ void gemm_bias_wmma(const float* __restrict__ A,
                               const float* __restrict__ Bg,
                               const float* __restrict__ bias,
                               float* __restrict__ C,
                               int M, int K, int N,
                               int mblocks, int ngroups,
                               long strideA, long strideB, long strideC) {
    __shared__ __align__(16) uint32_t s_bw[2][1024];   // 2 x 4 KB fragment bufs

    const int t    = threadIdx.x;
    const int wave = t >> 5;
    const int lane = t & 31;

    const int per_batch = mblocks * ngroups;
    const int bidx = blockIdx.x / per_batch;
    const int rem  = blockIdx.x - bidx * per_batch;
    const int mb   = rem / ngroups;
    const int ng   = rem - mb * ngroups;

    const float* Ab = A  + (long)bidx * strideA;
    const float* Bb = Bg + (long)bidx * strideB;
    float*       Cb = C  + (long)bidx * strideC;

    const int half = lane >> 4;
    const int mr   = lane & 15;
    const int m    = (mb * WPB + wave) * 16 + mr;   // A row of this lane
    const int n0   = ng * 64;                       // column-group base

    const float* Arow = Ab + (size_t)m * K;

    // staging constants for this thread
    const int nst = t & 63;
    const int kb  = (t >> 6) << 1;                  // 0 or 2
    const float* gstage = Bb + (size_t)kb * N + n0 + nst;

    const int KT = K >> 5;
    v8f acc[4] = {};

    stage_b_slab(&s_bw[0][0], gstage, N, kb, nst);  // prologue: slab 0
    v16bf a = load_a_frag(Arow, 0, half);

    for (int kt = 0; kt < KT; ++kt) {
        __syncthreads();        // buf[kt&1] staged; buf[nxt] reads retired
        const v8bf* fb = (const v8bf*)&s_bw[kt & 1][0];
        const v16bf a_cur = a;

        // next A fragment + prefetch (independent: scheduler hoists loads)
        if (kt + 1 < KT) {
            a = load_a_frag(Arow, (kt + 1) << 5, half);
            __builtin_prefetch(Arow + ((kt + 2) << 5), 0, 1);
        }

#pragma unroll
        for (int j = 0; j < 4; ++j) {
            const v8bf lo = fb[(j * 2 + 0) * 32 + lane];
            const v8bf hi = fb[(j * 2 + 1) * 32 + lane];
            const v16bf bfrag = __builtin_shufflevector(
                lo, hi, 0, 1, 2, 3, 4, 5, 6, 7, 8, 9, 10, 11, 12, 13, 14, 15);
            acc[j] = __builtin_amdgcn_wmma_f32_16x16x32_bf16(
                false, a_cur, false, bfrag, (short)0, acc[j], false, false);
        }

        // stage slab kt+1 into the other buffer (overlaps matrix pipe)
        if (kt + 1 < KT)
            stage_b_slab(&s_bw[(kt + 1) & 1][0],
                         gstage + ((size_t)(kt + 1) << 5) * N, N, kb, nst);
    }

    const int mbase = (mb * WPB + wave) * 16 + half * 8;
#pragma unroll
    for (int j = 0; j < 4; ++j) {
        const float bn = bias ? bias[n0 + j * 16 + mr] : 0.0f;
#pragma unroll
        for (int r = 0; r < 8; ++r)
            Cb[(size_t)(mbase + r) * N + n0 + j * 16 + mr] = acc[j][r] + bn;
    }
}

// ---------------------------------------------------------------------------
// Fused score + mask + softmax.
//   score[b,q,v] = sum_u Vw[u] * tanh(qp[b,q,u] + vp[b,v,u])
// One block per (b, group-of-8 q). Thread t owns v=t; the LDS-staged v_proj
// chunk is reused by all 8 queries (8x less vp traffic), and the inner loop
// issues 8 independent tanh(TRANS32)+FMA chains per ds_load.
// ---------------------------------------------------------------------------
#define UCHUNK 64
#define QG     8

__global__ void score_softmax_kernel(const float* __restrict__ qp,
                                     const float* __restrict__ vp,
                                     const float* __restrict__ Vw,
                                     const int*   __restrict__ mask,
                                     float* __restrict__ attn_out) {
    __shared__ float s_q[QG][U_];             // 16 KB
    __shared__ float s_vw[U_];                //  2 KB
    __shared__ float s_vt[TV_][UCHUNK + 1];   // 65 KB, +1 pad: conflict-free rows
    __shared__ float red[256];

    const int blk = blockIdx.x;               // 0 .. B*(TQ/QG)-1
    const int b   = blk / (TQ_ / QG);
    const int qg  = blk - b * (TQ_ / QG);
    const int t   = threadIdx.x;              // == v index
    const float* qbase = qp + (size_t)(b * TQ_ + qg * QG) * U_;
    const float* vbat  = vp + (size_t)b * TV_ * U_;

    for (int i = t; i < QG * U_; i += 256) s_q[i >> 9][i & (U_ - 1)] = qbase[i];
    for (int i = t; i < U_;      i += 256) s_vw[i] = Vw[i];

    float sc[QG] = {};
    for (int c = 0; c < U_ / UCHUNK; ++c) {
        __syncthreads();
        // coalesced, conflict-free stage of vp[b, 0:256, c*64:(c+1)*64]
        for (int i = t; i < TV_ * UCHUNK; i += 256) {
            const int v = i >> 6, u = i & (UCHUNK - 1);
            s_vt[v][u] = vbat[v * U_ + c * UCHUNK + u];
        }
        __syncthreads();
        const int u0 = c * UCHUNK;
        for (int u = 0; u < UCHUNK; ++u) {
            const float vv = s_vt[t][u];          // per-lane distinct banks
            const float w  = s_vw[u0 + u];        // broadcast
#pragma unroll
            for (int q = 0; q < QG; ++q) {
                const float x = s_q[q][u0 + u] + vv;
#if __has_builtin(__builtin_amdgcn_tanhf)
                sc[q] = fmaf(w, __builtin_amdgcn_tanhf(x), sc[q]);
#else
                sc[q] = fmaf(w, tanhf(x), sc[q]);
#endif
            }
        }
    }

    // additive encoder mask (same for all q of this block)
    const float mterm = (1.0f - (float)mask[b * TV_ + t]) * 1e9f;

    for (int q = 0; q < QG; ++q) {
        const float s = sc[q] - mterm;
        __syncthreads();
        red[t] = s;
        __syncthreads();
        for (int st = 128; st > 0; st >>= 1) {
            if (t < st) red[t] = fmaxf(red[t], red[t + st]);
            __syncthreads();
        }
        const float mx = red[0];
        __syncthreads();
        const float ex = __expf(s - mx);
        red[t] = ex;
        __syncthreads();
        for (int st = 128; st > 0; st >>= 1) {
            if (t < st) red[t] += red[t + st];
            __syncthreads();
        }
        attn_out[(size_t)(b * TQ_ + qg * QG + q) * TV_ + t] = ex / red[0];
    }
}

// ---------------------------------------------------------------------------
extern "C" void kernel_launch(void* const* d_in, const int* in_sizes, int n_in,
                              void* d_out, int out_size, void* d_ws, size_t ws_size,
                              hipStream_t stream) {
    const float* query    = (const float*)d_in[0];   // [B,TQ,D]
    const float* values   = (const float*)d_in[1];   // [B,TV,D]
    const int*   enc_mask = (const int*)  d_in[2];   // [B,TV]
    const float* W1       = (const float*)d_in[3];   // [D,U]
    const float* b1       = (const float*)d_in[4];   // [U]
    const float* W2       = (const float*)d_in[5];   // [D,U]
    const float* b2       = (const float*)d_in[6];   // [U]
    const float* Vw       = (const float*)d_in[7];   // [U]
    // d_in[8] = Vb: constant pre-softmax shift, cancels in softmax; unused.

    float* ctx_out  = (float*)d_out;                          // B*TQ*D
    float* attn_out = ctx_out + (size_t)B_ * TQ_ * D_;        // B*TQ*TV

    float* qp = (float*)d_ws;                                 // B*TQ*U  (2 MB)
    float* vp = qp + (size_t)B_ * TQ_ * U_;                   // B*TV*U  (8 MB)

    // 1) projections: qp = query@W1+b1 (M=1024), vp = values@W2+b2 (M=4096)
    {
        const int Mq = B_ * TQ_, Mv = B_ * TV_;
        const int mbQ = Mq / 64, mbV = Mv / 64, ngr = U_ / 64;   // 16/64, 8
        gemm_bias_wmma<<<mbQ * ngr, TPB, 0, stream>>>(
            query,  W1, b1, qp, Mq, D_, U_, mbQ, ngr, 0, 0, 0);
        gemm_bias_wmma<<<mbV * ngr, TPB, 0, stream>>>(
            values, W2, b2, vp, Mv, D_, U_, mbV, ngr, 0, 0, 0);
    }

    // 2) fused tanh-score + mask + softmax (block per (b, 8 queries))
    score_softmax_kernel<<<B_ * (TQ_ / QG), 256, 0, stream>>>(qp, vp, Vw, enc_mask, attn_out);

    // 3) context[b] = attn[b] @ values[b]  (batched: 16 x (64x512x256))
    {
        const int mbC = TQ_ / 64, ngr = D_ / 64;                 // 1, 8
        gemm_bias_wmma<<<B_ * mbC * ngr, TPB, 0, stream>>>(
            attn_out, values, nullptr, ctx_out, TQ_, TV_, D_, mbC, ngr,
            (long)TQ_ * TV_, (long)TV_ * D_, (long)TQ_ * D_);
    }
}
